// GraphPool_18013092840066
// MI455X (gfx1250) — matrix operand order repaired and checked by reading, not compiled, not run on hardware
//
#include <hip/hip_runtime.h>

typedef __attribute__((ext_vector_type(16))) _Float16 v16h;
typedef __attribute__((ext_vector_type(8)))  float    v8f;

#define N_NODES 32768
#define D_FEAT  64
#define E_EDGES 524288
#define NS_SEL  8192
#define K_NBR   16
#define R_CUT   0.15f
#define EPS_RMS 1e-6f
#define LDK_M   160   // padded K for edge MLP (2*64+4=132 -> 160, 5 chunks of 32)

// ---------- WMMA fragment loaders (wave32, 16x16x32 f16) ----------
// A (16x32): lane half h holds k = h*16+i (contiguous); row = lane%16
static __device__ __forceinline__ v16h load_a16(const _Float16* rowbase, int kb, int half) {
  const _Float16* p0 = rowbase + kb + half * 16;
  v16h a;
#pragma unroll
  for (int i = 0; i < 16; i++) { a[i] = p0[i]; }
  return a;
}
// B (32x16): lane holds col n = lane%16; k = half*16 + i (contiguous when stored transposed)
static __device__ __forceinline__ v16h load_b16(const _Float16* WT, int n, int ldk, int kb, int half) {
  const _Float16* p = WT + n * ldk + kb + half * 16;
  v16h b;
#pragma unroll
  for (int i = 0; i < 16; i++) b[i] = p[i];
  return b;
}
// cooperative block-wide copy of f16 weights (global, L2-hot) -> LDS, 16B vectors
static __device__ __forceinline__ void stage_weights(const _Float16* __restrict__ g,
                                                     _Float16* __restrict__ l, int halves) {
  const uint4* s4 = (const uint4*)g;
  uint4* d4 = (uint4*)l;
  const int n4 = halves >> 3;
  for (int i = threadIdx.x; i < n4; i += blockDim.x) d4[i] = s4[i];
}

// ---------- small prep kernels ----------
__global__ void f32_to_f16_kernel(const float* __restrict__ src, _Float16* __restrict__ dst, int n) {
  int t = blockIdx.x * blockDim.x + threadIdx.x;
  if (t < n) dst[t] = (_Float16)src[t];
}

// dst[n*ldk + k] = (k<K) ? src[k*Nc + n] : 0   (transpose + f16, zero-padded K)
__global__ void convertT_kernel(const float* __restrict__ src, _Float16* __restrict__ dst,
                                int K, int Nc, int ldk) {
  int t = blockIdx.x * blockDim.x + threadIdx.x;
  if (t >= Nc * ldk) return;
  int n = t / ldk, k = t % ldk;
  dst[t] = (k < K) ? (_Float16)src[k * Nc + n] : (_Float16)0.0f;
}

__global__ void zero_kernel(float* __restrict__ p, int n) {
  int t = blockIdx.x * blockDim.x + threadIdx.x;
  if (t < n) p[t] = 0.0f;
}

// ---------- edge message-passing: m = relu([hn[s],hn[d],rel,d] @ Wm + bm); agg[dst]+=m ----------
__global__ __launch_bounds__(256) void edge_mp_kernel(
    const _Float16* __restrict__ hnf16, const float* __restrict__ pos,
    const int* __restrict__ srcIdx, const int* __restrict__ dstIdx,
    const _Float16* __restrict__ WmT, const float* __restrict__ bm,
    float* __restrict__ agg)
{
  __shared__ _Float16 sm[8][16 * LDK_M];     // 40 KB: 8 waves x 16 edges x 160 f16
  __shared__ _Float16 swt[64 * LDK_M];       // 20 KB: weights staged once per block
  __shared__ int sdst[8][16];
  const int wave = threadIdx.x >> 5;
  const int lane = threadIdx.x & 31;
  const int half = lane >> 4;
  const int l16  = lane & 15;
  const int eb   = blockIdx.x * 128 + wave * 16;
  const int e    = eb + l16;
  const int s    = srcIdx[e];
  const int d    = dstIdx[e];
  _Float16* row  = &sm[wave][l16 * LDK_M];

  stage_weights(WmT, swt, 64 * LDK_M);       // B matrix: global -> LDS once per block

  { // stage 64 f16 node features (src half or dst half) as 8x uint4 (128B, L2-resident gather)
    const int node = half ? d : s;
    const uint4* hp = (const uint4*)(hnf16 + node * D_FEAT);
    uint4* dp = (uint4*)(row + half * 64);
#pragma unroll
    for (int i = 0; i < 8; i++) dp[i] = hp[i];
  }
  if (half) {
    float rx = pos[d * 3 + 0] - pos[s * 3 + 0];
    float ry = pos[d * 3 + 1] - pos[s * 3 + 1];
    float rz = pos[d * 3 + 2] - pos[s * 3 + 2];
    float dd = sqrtf(rx * rx + ry * ry + rz * rz);
    row[128] = (_Float16)rx; row[129] = (_Float16)ry;
    row[130] = (_Float16)rz; row[131] = (_Float16)dd;
  } else {
    sdst[wave][l16] = d;
#pragma unroll
    for (int i = 132; i < LDK_M; i++) row[i] = (_Float16)0.0f;   // zero K padding
  }
  __syncthreads();

  v8f acc[4];
#pragma unroll
  for (int nc = 0; nc < 4; nc++) {
    float bv = bm[l16 + nc * 16];            // bias pre-loaded into accumulator
#pragma unroll
    for (int i = 0; i < 8; i++) acc[nc][i] = bv;
  }
  const _Float16* arow = &sm[wave][l16 * LDK_M];
#pragma unroll
  for (int kc = 0; kc < 5; kc++) {           // K = 160 = 5 x 32
    const int kb = kc * 32;
    v16h a = load_a16(arow, kb, half);
#pragma unroll
    for (int nc = 0; nc < 4; nc++) {         // N = 64 = 4 x 16
      v16h b = load_b16(swt, l16 + nc * 16, LDK_M, kb, half);
      acc[nc] = __builtin_amdgcn_wmma_f32_16x16x32_f16(
          false, a, false, b, (short)0, acc[nc], false, false);
    }
  }
  // ReLU + segment_sum via L2-resident f32 atomics
#pragma unroll
  for (int nc = 0; nc < 4; nc++) {
    const int col = l16 + nc * 16;
#pragma unroll
    for (int r = 0; r < 8; r++) {
      float v = acc[nc][r];
      v = v > 0.0f ? v : 0.0f;
      const int rowe = r + 8 * half;         // C/D layout: lanes<16 rows 0-7, lanes>=16 rows 8-15
      atomicAdd(&agg[sdst[wave][rowe] * D_FEAT + col], v);
    }
  }
}

// ---------- node update: rms_norm(hn + agg @ Wu) ----------
__global__ __launch_bounds__(256) void node_update_kernel(
    const float* __restrict__ hn_in, const float* __restrict__ agg,
    const _Float16* __restrict__ WuT,
    float* __restrict__ hn_out, _Float16* __restrict__ hn_out16)
{
  __shared__ _Float16 sa[8][16 * D_FEAT];    // 16 KB
  __shared__ _Float16 swt[64 * D_FEAT];      // 8 KB weights
  const int wave = threadIdx.x >> 5, lane = threadIdx.x & 31;
  const int half = lane >> 4, l16 = lane & 15;
  const int nb = blockIdx.x * 128 + wave * 16;

  stage_weights(WuT, swt, 64 * D_FEAT);

  { // stage agg tile as f16
    const float* src = agg + (nb + l16) * D_FEAT + half * 32;
    _Float16* dst = &sa[wave][l16 * D_FEAT + half * 32];
#pragma unroll
    for (int i = 0; i < 32; i++) dst[i] = (_Float16)src[i];
  }
  __syncthreads();

  v8f acc[4];
#pragma unroll
  for (int nc = 0; nc < 4; nc++)
#pragma unroll
    for (int i = 0; i < 8; i++) acc[nc][i] = 0.0f;

  const _Float16* arow = &sa[wave][l16 * D_FEAT];
#pragma unroll
  for (int kc = 0; kc < 2; kc++) {
    const int kb = kc * 32;
    v16h a = load_a16(arow, kb, half);
#pragma unroll
    for (int nc = 0; nc < 4; nc++) {
      v16h b = load_b16(swt, l16 + nc * 16, D_FEAT, kb, half);
      acc[nc] = __builtin_amdgcn_wmma_f32_16x16x32_f16(
          false, a, false, b, (short)0, acc[nc], false, false);
    }
  }
  // residual + RMS norm; row reduction across the 16 lanes of each half via shfl_xor
  float v[4][8];
#pragma unroll
  for (int r = 0; r < 8; r++) {
    const int node = nb + r + 8 * half;
    float p = 0.0f;
#pragma unroll
    for (int nc = 0; nc < 4; nc++) {
      const int col = l16 + nc * 16;
      float x = hn_in[node * D_FEAT + col] + acc[nc][r];
      v[nc][r] = x;
      p += x * x;
    }
    p += __shfl_xor(p, 1, 32);
    p += __shfl_xor(p, 2, 32);
    p += __shfl_xor(p, 4, 32);
    p += __shfl_xor(p, 8, 32);               // bit4 untouched -> stays within half
    const float scale = rsqrtf(p * (1.0f / 64.0f) + EPS_RMS);
#pragma unroll
    for (int nc = 0; nc < 4; nc++) {
      const int col = l16 + nc * 16;
      float o = v[nc][r] * scale;
      hn_out[node * D_FEAT + col] = o;
      hn_out16[node * D_FEAT + col] = (_Float16)o;
    }
  }
}

// ---------- scoring: s = sum_c (hn@W_lin)_c^2 * w_tp_c; pack sortable u64 key ----------
__global__ void score_kernel(const float* __restrict__ hn, const float* __restrict__ Wlin,
                             const float* __restrict__ wtp, unsigned long long* __restrict__ keys)
{
  const int n = blockIdx.x * blockDim.x + threadIdx.x;
  if (n >= N_NODES) return;
  float h[64];
#pragma unroll
  for (int k = 0; k < 64; k++) h[k] = hn[n * 64 + k];
  float s = 0.0f;
#pragma unroll
  for (int c = 0; c < 16; c++) {
    float sc = 0.0f;
#pragma unroll
    for (int k = 0; k < 64; k++) sc += h[k] * Wlin[k * 16 + c];
    s += sc * sc * wtp[c];
  }
  unsigned int b = __float_as_uint(s);
  unsigned int m = (b & 0x80000000u) ? ~b : (b | 0x80000000u);  // monotonic float->uint
  keys[n] = (((unsigned long long)m) << 32) | (unsigned int)n;
}

// ---------- exact top-NS: global bitonic sort (descending) ----------
__global__ void bitonic_pass(unsigned long long* __restrict__ keys, int j, int k) {
  const int i = blockIdx.x * blockDim.x + threadIdx.x;
  const int l = i ^ j;
  if (l > i) {
    unsigned long long a = keys[i], b = keys[l];
    const bool desc = ((i & k) == 0);
    if (desc ? (a < b) : (a > b)) { keys[i] = b; keys[l] = a; }
  }
}

// ---------- gather selected nodes, gate, precompute |pos|^2 ----------
__global__ void gather_kernel(const unsigned long long* __restrict__ keys,
                              const float* __restrict__ hn, const float* __restrict__ pos,
                              float* __restrict__ hn_s_out, float* __restrict__ pos_s_out,
                              float* __restrict__ pos_s_ws, float* __restrict__ sq_ws)
{
  const int i = blockIdx.x * blockDim.x + threadIdx.x;
  if (i >= NS_SEL) return;
  const unsigned long long key = keys[i];
  const int idx = (int)(key & 0xffffffffu);
  const unsigned int m = (unsigned int)(key >> 32);
  const unsigned int b = (m & 0x80000000u) ? (m ^ 0x80000000u) : ~m;
  const float sval = __uint_as_float(b);
  const float gate = 1.0f / (1.0f + expf(-sval));
  for (int c = 0; c < 64; c++) hn_s_out[i * 64 + c] = hn[idx * 64 + c] * gate;
  const float px = pos[idx * 3], py = pos[idx * 3 + 1], pz = pos[idx * 3 + 2];
  pos_s_out[i * 3] = px; pos_s_out[i * 3 + 1] = py; pos_s_out[i * 3 + 2] = pz;
  pos_s_ws[i * 3] = px; pos_s_ws[i * 3 + 1] = py; pos_s_ws[i * 3 + 2] = pz;
  sq_ws[i] = px * px + py * py + pz * pz;
}

// ---------- brute-force kNN with LDS tiling + register top-17 ----------
#define TP 512
__global__ __launch_bounds__(256) void knn_kernel(
    const float* __restrict__ pos_s, const float* __restrict__ sq,
    int* __restrict__ nbr, float* __restrict__ dsel2,
    float* __restrict__ out_src, float* __restrict__ out_dst)
{
  __shared__ float tps[TP * 3];
  __shared__ float tsq[TP];
  const int row = blockIdx.x * blockDim.x + threadIdx.x;
  const float px = pos_s[row * 3], py = pos_s[row * 3 + 1], pz = pos_s[row * 3 + 2];
  const float sqi = sq[row];
  float bd[K_NBR + 1]; int bi[K_NBR + 1];
#pragma unroll
  for (int t = 0; t <= K_NBR; t++) { bd[t] = 3.0e38f; bi[t] = -1; }

  for (int tile = 0; tile < NS_SEL / TP; ++tile) {
    __syncthreads();
    for (int t = threadIdx.x; t < TP; t += blockDim.x) {
      const int g = tile * TP + t;
      tps[t * 3 + 0] = pos_s[g * 3 + 0];
      tps[t * 3 + 1] = pos_s[g * 3 + 1];
      tps[t * 3 + 2] = pos_s[g * 3 + 2];
      tsq[t] = sq[g];
    }
    __syncthreads();
    for (int t = 0; t < TP; t++) {
      const float dot = px * tps[t * 3] + py * tps[t * 3 + 1] + pz * tps[t * 3 + 2];
      const float d2 = sqi + tsq[t] - 2.0f * dot;
      if (d2 < bd[K_NBR]) {                  // register insertion, constant indices (no scratch)
        float cd = d2; int ci = tile * TP + t;
#pragma unroll
        for (int p = 0; p <= K_NBR; p++) {
          if (cd < bd[p]) {
            float td = bd[p]; int ti = bi[p];
            bd[p] = cd; bi[p] = ci; cd = td; ci = ti;
          }
        }
      }
    }
  }
  // entry 0 is self; keep next K neighbors
#pragma unroll
  for (int t = 1; t <= K_NBR; t++) {
    const int e = row * K_NBR + (t - 1);
    const float dd = bd[t] > 0.0f ? bd[t] : 0.0f;
    nbr[e] = bi[t]; dsel2[e] = dd;
    out_src[e] = (float)bi[t]; out_dst[e] = (float)row;
  }
}

// ---------- pooled-edge MLP: he = relu(fe@We1+be1)@We2, masked ----------
__global__ __launch_bounds__(256) void he_kernel(
    const float* __restrict__ pos_s, const int* __restrict__ nbr,
    const float* __restrict__ dsel2,
    const float* __restrict__ We1, const float* __restrict__ be1,
    const _Float16* __restrict__ We2T,
    float* __restrict__ he_out)
{
  __shared__ _Float16 sh1[8][16 * 64];       // 16 KB
  __shared__ _Float16 swt[64 * 64];          // 8 KB weights
  __shared__ float sfe[8][16 * 4];
  __shared__ float smask[8][16];
  const int wave = threadIdx.x >> 5, lane = threadIdx.x & 31;
  const int half = lane >> 4, l16 = lane & 15;
  const int eb = blockIdx.x * 128 + wave * 16;

  stage_weights(We2T, swt, 64 * 64);

  if (!half) {
    const int e = eb + l16;
    const int i = e / K_NBR;
    const int j = nbr[e];
    const float d2 = dsel2[e];
    const float mask = (d2 <= R_CUT * R_CUT) ? 1.0f : 0.0f;
    const float dist = sqrtf(d2 + 1e-12f);
    const float rx = pos_s[j * 3 + 0] - pos_s[i * 3 + 0];
    const float ry = pos_s[j * 3 + 1] - pos_s[i * 3 + 1];
    const float rz = pos_s[j * 3 + 2] - pos_s[i * 3 + 2];
    sfe[wave][l16 * 4 + 0] = rx * mask;
    sfe[wave][l16 * 4 + 1] = ry * mask;
    sfe[wave][l16 * 4 + 2] = rz * mask;
    sfe[wave][l16 * 4 + 3] = dist * mask;
    smask[wave][l16] = mask;
  }
  __syncthreads();

  // first layer (K=4): 1024 outputs per wave, 32 per lane, scalar VALU
  for (int t = 0; t < 32; t++) {
    const int o = t * 32 + lane;             // 0..1023 unique
    const int e = o >> 6, c = o & 63;
    float h = be1[c];
#pragma unroll
    for (int f = 0; f < 4; f++) h += sfe[wave][e * 4 + f] * We1[f * 64 + c];
    h = h > 0.0f ? h : 0.0f;
    sh1[wave][e * 64 + c] = (_Float16)h;
  }
  __syncthreads();

  v8f acc[4];
#pragma unroll
  for (int nc = 0; nc < 4; nc++)
#pragma unroll
    for (int i = 0; i < 8; i++) acc[nc][i] = 0.0f;

  const _Float16* arow = &sh1[wave][l16 * 64];
#pragma unroll
  for (int kc = 0; kc < 2; kc++) {
    const int kb = kc * 32;
    v16h a = load_a16(arow, kb, half);
#pragma unroll
    for (int nc = 0; nc < 4; nc++) {
      v16h b = load_b16(swt, l16 + nc * 16, 64, kb, half);
      acc[nc] = __builtin_amdgcn_wmma_f32_16x16x32_f16(
          false, a, false, b, (short)0, acc[nc], false, false);
    }
  }
#pragma unroll
  for (int nc = 0; nc < 4; nc++) {
    const int col = l16 + nc * 16;
#pragma unroll
    for (int r = 0; r < 8; r++) {
      const int rowe = r + 8 * half;
      const float mv = smask[wave][rowe];
      he_out[(eb + rowe) * 64 + col] = acc[nc][r] * mv;
    }
  }
}

// ================= host-side orchestration =================
extern "C" void kernel_launch(void* const* d_in, const int* in_sizes, int n_in,
                              void* d_out, int out_size, void* d_ws, size_t ws_size,
                              hipStream_t stream)
{
  (void)in_sizes; (void)n_in; (void)out_size; (void)ws_size;
  const float* hn   = (const float*)d_in[0];
  const float* pos  = (const float*)d_in[1];
  const int*   eidx = (const int*)d_in[2];
  const int*   esrc = eidx;
  const int*   edst = eidx + E_EDGES;
  const float* Wm1  = (const float*)d_in[3];
  const float* bm1  = (const float*)d_in[4];
  const float* Wu1  = (const float*)d_in[5];
  const float* Wm2  = (const float*)d_in[6];
  const float* bm2  = (const float*)d_in[7];
  const float* Wu2  = (const float*)d_in[8];
  const float* Wlin = (const float*)d_in[9];
  const float* wtp  = (const float*)d_in[10];
  const float* We1  = (const float*)d_in[11];
  const float* be1  = (const float*)d_in[12];
  const float* We2  = (const float*)d_in[13];

  char* ws = (char*)d_ws;
  size_t off = 0;
  auto alloc = [&](size_t bytes) { size_t r = off; off += (bytes + 255) & ~size_t(255); return r; };
  _Float16* hn0_16 = (_Float16*)(ws + alloc((size_t)N_NODES * 64 * 2));
  float*    hn1    = (float*)   (ws + alloc((size_t)N_NODES * 64 * 4));
  _Float16* hn1_16 = (_Float16*)(ws + alloc((size_t)N_NODES * 64 * 2));
  float*    hn2    = (float*)   (ws + alloc((size_t)N_NODES * 64 * 4));
  float*    agg    = (float*)   (ws + alloc((size_t)N_NODES * 64 * 4));
  _Float16* Wm1T   = (_Float16*)(ws + alloc(64 * LDK_M * 2));
  _Float16* Wu1T   = (_Float16*)(ws + alloc(64 * 64 * 2));
  _Float16* Wm2T   = (_Float16*)(ws + alloc(64 * LDK_M * 2));
  _Float16* Wu2T   = (_Float16*)(ws + alloc(64 * 64 * 2));
  _Float16* We2T   = (_Float16*)(ws + alloc(64 * 64 * 2));
  unsigned long long* keys = (unsigned long long*)(ws + alloc((size_t)N_NODES * 8));
  float* pos_s = (float*)(ws + alloc((size_t)NS_SEL * 3 * 4));
  float* sqv   = (float*)(ws + alloc((size_t)NS_SEL * 4));
  int*   nbr   = (int*)  (ws + alloc((size_t)NS_SEL * K_NBR * 4));
  float* dse2  = (float*)(ws + alloc((size_t)NS_SEL * K_NBR * 4));
  _Float16* hn2_16 = (_Float16*)(ws + alloc((size_t)N_NODES * 64 * 2)); // scratch sink

  float* out      = (float*)d_out;
  float* out_hns  = out;                                   // NS x 64
  float* out_poss = out_hns + (size_t)NS_SEL * 64;         // NS x 3
  float* out_src  = out_poss + (size_t)NS_SEL * 3;         // NS*K
  float* out_dst  = out_src + (size_t)NS_SEL * K_NBR;      // NS*K
  float* out_he   = out_dst + (size_t)NS_SEL * K_NBR;      // NS*K x 64

  const int T = 256;
  // ---- prep: f16 copies of hn and transposed f16 weights ----
  f32_to_f16_kernel<<<(N_NODES * 64 + T - 1) / T, T, 0, stream>>>(hn, hn0_16, N_NODES * 64);
  convertT_kernel<<<(64 * LDK_M + T - 1) / T, T, 0, stream>>>(Wm1, Wm1T, 132, 64, LDK_M);
  convertT_kernel<<<(64 * 64 + T - 1) / T, T, 0, stream>>>(Wu1, Wu1T, 64, 64, 64);
  convertT_kernel<<<(64 * LDK_M + T - 1) / T, T, 0, stream>>>(Wm2, Wm2T, 132, 64, LDK_M);
  convertT_kernel<<<(64 * 64 + T - 1) / T, T, 0, stream>>>(Wu2, Wu2T, 64, 64, 64);
  convertT_kernel<<<(64 * 64 + T - 1) / T, T, 0, stream>>>(We2, We2T, 64, 64, 64);

  // ---- message-passing block 1 ----
  zero_kernel<<<(N_NODES * 64 + T - 1) / T, T, 0, stream>>>(agg, N_NODES * 64);
  edge_mp_kernel<<<E_EDGES / 128, T, 0, stream>>>(hn0_16, pos, esrc, edst, Wm1T, bm1, agg);
  node_update_kernel<<<N_NODES / 128, T, 0, stream>>>(hn, agg, Wu1T, hn1, hn1_16);

  // ---- message-passing block 2 ----
  zero_kernel<<<(N_NODES * 64 + T - 1) / T, T, 0, stream>>>(agg, N_NODES * 64);
  edge_mp_kernel<<<E_EDGES / 128, T, 0, stream>>>(hn1_16, pos, esrc, edst, Wm2T, bm2, agg);
  node_update_kernel<<<N_NODES / 128, T, 0, stream>>>(hn1, agg, Wu2T, hn2, hn2_16);

  // ---- scoring + exact descending bitonic sort of 32768 keys ----
  score_kernel<<<N_NODES / T, T, 0, stream>>>(hn2, Wlin, wtp, keys);
  for (int k = 2; k <= N_NODES; k <<= 1)
    for (int j = k >> 1; j > 0; j >>= 1)
      bitonic_pass<<<N_NODES / T, T, 0, stream>>>(keys, j, k);

  // ---- pooled graph construction ----
  gather_kernel<<<NS_SEL / T, T, 0, stream>>>(keys, hn2, pos, out_hns, out_poss, pos_s, sqv);
  knn_kernel<<<NS_SEL / T, T, 0, stream>>>(pos_s, sqv, nbr, dse2, out_src, out_dst);
  he_kernel<<<(NS_SEL * K_NBR) / 128, T, 0, stream>>>(pos_s, nbr, dse2, We1, be1, We2T, out_he);
}
